// CascadedAttention_25228637897001
// MI455X (gfx1250) — compile-verified, matrix operand order checked by reading, not run on hardware
//
#include <hip/hip_runtime.h>
#include <hip/hip_bf16.h>
#include <math.h>

#define NB 32
#define NT 1024
#define ND 1024
#define NO 256

typedef __attribute__((ext_vector_type(16))) __bf16 bf16x16;
typedef __attribute__((ext_vector_type(8)))  float   f32x8;

union Frag16 {
  bf16x16 v;
  uint4 q[2];
  unsigned short s[16];
};

__device__ __forceinline__ unsigned short f2bf(float x){
  unsigned int u = __float_as_uint(x);
  unsigned int r = u + 0x7FFFu + ((u >> 16) & 1u);   // round-to-nearest-even
  return (unsigned short)(r >> 16);
}
__device__ __forceinline__ float bf2f_lo(unsigned int x){       // bf16 in low half
  return __uint_as_float(x << 16);
}
__device__ __forceinline__ float bf2f_hi(unsigned int x){       // bf16 in high half
  return __uint_as_float(x & 0xFFFF0000u);
}
__device__ __forceinline__ float fast_tanh(float x){
  float r;
  asm("v_tanh_f32 %0, %1" : "=v"(r) : "v"(x));                  // gfx1250 TRANS op
  return r;
}
__device__ __forceinline__ f32x8 zero8(){
  f32x8 z;
#pragma unroll
  for (int i = 0; i < 8; ++i) z[i] = 0.f;
  return z;
}

// ---------------------------------------------------------------- prep kernels

// XbfT[b, d, t] = bf16(inputs[b, t, d])   (tiled 32x32 LDS transpose)
__global__ void k_transpose_bf(const float* __restrict__ in, unsigned short* __restrict__ out){
  __shared__ unsigned short tile[32][33];
  int b  = blockIdx.z;
  int t0 = blockIdx.x * 32;
  int d0 = blockIdx.y * 32;
  int lx = threadIdx.x & 31;
  int ly = threadIdx.x >> 5;        // 0..7
#pragma unroll
  for (int r = 0; r < 32; r += 8){
    int t = t0 + ly + r;
    tile[ly + r][lx] = f2bf(in[((long long)b * NT + t) * ND + d0 + lx]);
  }
  __syncthreads();
#pragma unroll
  for (int r = 0; r < 32; r += 8){
    int d = d0 + ly + r;
    out[((long long)b * ND + d) * NT + t0 + lx] = tile[lx][ly + r];
  }
}

// Pack a row-major [K,N] fp32 weight into bf16 B-fragment layout (ISA 16-bit layout).
__global__ void k_pack_b(const float* __restrict__ W, unsigned short* __restrict__ out, int K, int N){
  int total = K * N;
  int ntiles = N >> 4;
  for (int idx = blockIdx.x * blockDim.x + threadIdx.x; idx < total; idx += gridDim.x * blockDim.x){
    int e    = idx & 15;
    int lane = (idx >> 4) & 31;
    int tile = idx >> 9;
    int nt = tile % ntiles;
    int kt = tile / ntiles;
    int half = lane >> 4;
    int n = nt * 16 + (lane & 15);
    int v = e >> 1, p = e & 1;
    int kin = (v < 4 ? v * 2 : 16 + (v - 4) * 2) + half * 8 + p;
    int k = kt * 32 + kin;
    out[idx] = f2bf(W[(long long)k * N + n]);
  }
}

// embWo[o] = sum_j emb[o,j] * Wo[j]   (hoists (softmax(p)@emb)@Wo -> softmax(p)·embWo)
__global__ void k_embwo(const float* __restrict__ emb, const float* __restrict__ Wo,
                        float* __restrict__ embWo){
  int o = threadIdx.x;
  float s = 0.f;
  for (int j = 0; j < NO; ++j) s += emb[(long long)o * NO + j] * Wo[j];
  embWo[o] = s;
}

// ------------------------------------------------- big pre-GEMMs (WMMA bf16)
// out[B*T, O] = A[B*T, D] @ Wpack + bias.  Wave owns 16 rows x 8 N-tiles
// (64 accumulator VGPRs -> no spills).  A converted fp32->bf16 on the fly.
__global__ void __launch_bounds__(256)
k_gemm_bto(const float* __restrict__ Xf,
           const unsigned short* __restrict__ Wpack,
           const float* __restrict__ bias, int biasPerT,
           float* __restrict__ outp, int rollPrev){
  int lane = threadIdx.x & 31;
  int wave = blockIdx.x * (blockDim.x >> 5) + (threadIdx.x >> 5); // 0..4095
  int mtile = wave >> 1;         // 0..2047
  int cg    = wave & 1;          // column group: N-tiles [cg*8, cg*8+8)
  int half = lane >> 4;
  int mloc = lane & 15;
  int mglob = mtile * 16 + mloc;
  int bidx = mglob >> 10;
  int tt   = mglob & (NT - 1);
  int srcT = rollPrev ? ((tt + NT - 1) & (NT - 1)) : tt;
  const float* Arow = Xf + ((long long)(bidx * NT + srcT)) * ND;

  f32x8 acc[8];
#pragma unroll
  for (int nt = 0; nt < 8; ++nt) acc[nt] = zero8();

  for (int kt = 0; kt < ND / 32; ++kt){
    Frag16 a;
    {
      const float* p0 = Arow + kt * 32 + half * 8;
      float4 f0 = *(const float4*)(p0);
      float4 f1 = *(const float4*)(p0 + 4);
      float4 f2 = *(const float4*)(p0 + 16);
      float4 f3 = *(const float4*)(p0 + 20);
      a.s[0] = f2bf(f0.x); a.s[1] = f2bf(f0.y); a.s[2] = f2bf(f0.z); a.s[3] = f2bf(f0.w);
      a.s[4] = f2bf(f1.x); a.s[5] = f2bf(f1.y); a.s[6] = f2bf(f1.z); a.s[7] = f2bf(f1.w);
      a.s[8] = f2bf(f2.x); a.s[9] = f2bf(f2.y); a.s[10] = f2bf(f2.z); a.s[11] = f2bf(f2.w);
      a.s[12] = f2bf(f3.x); a.s[13] = f2bf(f3.y); a.s[14] = f2bf(f3.z); a.s[15] = f2bf(f3.w);
    }
    const unsigned short* wk = Wpack + (((long long)kt * 16 + cg * 8) * 32 + lane) * 16;
#pragma unroll
    for (int nt = 0; nt < 8; ++nt){
      Frag16 bq;
      const uint4* bp = (const uint4*)(wk + (long long)nt * 32 * 16);
      bq.q[0] = bp[0];
      bq.q[1] = bp[1];
      acc[nt] = __builtin_amdgcn_wmma_f32_16x16x32_bf16(false, a.v, false, bq.v,
                                                        (short)0, acc[nt], false, false);
    }
  }
  // C/D layout: col = lane%16, row = v + 8*half
#pragma unroll
  for (int nt = 0; nt < 8; ++nt){
#pragma unroll
    for (int v = 0; v < 8; ++v){
      int row = mtile * 16 + v + 8 * half;
      int col = (cg * 8 + nt) * 16 + mloc;
      int trow = row & (NT - 1);
      float bv = biasPerT ? bias[(long long)trow * NO + col] : bias[col];
      outp[(long long)row * NO + col] = acc[nt][v] + bv;
    }
  }
}

// ------------------------------------------- per-step attention (1 block / batch)
__global__ void k_step_attn(const float* __restrict__ preds,        // d_out [B,T,O]
                            const float* __restrict__ Wa, const float* __restrict__ Va,
                            const float* __restrict__ Ba1, const float* __restrict__ Ba3,
                            const float* __restrict__ UaH,
                            const unsigned short* __restrict__ XbfT,  // [B,D,T]
                            const float* __restrict__ embWo,
                            unsigned short* __restrict__ ctx_bf, float* __restrict__ woy,
                            int t){
  __shared__ __attribute__((aligned(16))) float tanhp[NO];
  __shared__ __attribute__((aligned(16))) float was[NO];
  __shared__ __attribute__((aligned(16))) float vas[NO];
  __shared__ __attribute__((aligned(16))) float sc[NT];
  __shared__ __attribute__((aligned(16))) float red[NT];
  int b = blockIdx.x;
  int tid = threadIdx.x;

  // previous prediction (zeros at t==0); stage Va in LDS
  float p = 0.f;
  if (tid < NO && t > 0) p = preds[((long long)b * NT + (t - 1)) * NO + tid];
  if (tid < NO){ tanhp[tid] = fast_tanh(p); vas[tid] = Va[tid]; }

  // woy = softmax(pred) . embWo
  red[tid] = (tid < NO) ? p : -1e30f;
  __syncthreads();
  for (int s = NT / 2; s > 0; s >>= 1){ if (tid < s) red[tid] = fmaxf(red[tid], red[tid + s]); __syncthreads(); }
  float pmax = red[0];
  __syncthreads();
  float e = (tid < NO) ? __expf(p - pmax) : 0.f;
  red[tid] = e;
  __syncthreads();
  for (int s = NT / 2; s > 0; s >>= 1){ if (tid < s) red[tid] += red[tid + s]; __syncthreads(); }
  float esum = red[0];
  __syncthreads();
  red[tid] = (tid < NO) ? e * embWo[tid] : 0.f;
  __syncthreads();
  for (int s = NT / 2; s > 0; s >>= 1){ if (tid < s) red[tid] += red[tid + s]; __syncthreads(); }
  if (tid == 0) woy[b] = red[0] / esum;
  __syncthreads();

  // WaS = tanh(pred) @ Wa + Ba1  (4 threads per output column, LDS combine)
  {
    int o = tid & 255, part = tid >> 8;           // part: 0..3
    float acc = 0.f;
    const float* wcol = Wa + (long long)(part * 64) * NO + o;
    for (int k = 0; k < 64; ++k) acc += tanhp[part * 64 + k] * wcol[(long long)k * NO];
    red[tid] = acc;
  }
  __syncthreads();
  if (tid < NO) was[tid] = Ba1[tid] + red[tid] + red[tid + 256] + red[tid + 512] + red[tid + 768];
  __syncthreads();

  // scores[tt] = Va . tanh(UaH[b,tt,:] + WaS) + Ba3[tt]   (float4 + v_tanh)
  {
    const float* uah = UaH + ((long long)b * NT + tid) * NO;
    float s = Ba3[tid];
    for (int o = 0; o < NO; o += 4){
      float4 u = *(const float4*)(uah + o);
      float4 w = *(const float4*)&was[o];
      float4 v = *(const float4*)&vas[o];
      s += v.x * fast_tanh(u.x + w.x);
      s += v.y * fast_tanh(u.y + w.y);
      s += v.z * fast_tanh(u.z + w.z);
      s += v.w * fast_tanh(u.w + w.w);
    }
    sc[tid] = s;
  }
  __syncthreads();

  // softmax over T
  red[tid] = sc[tid]; __syncthreads();
  for (int s = NT / 2; s > 0; s >>= 1){ if (tid < s) red[tid] = fmaxf(red[tid], red[tid + s]); __syncthreads(); }
  float smax = red[0]; __syncthreads();
  float ee = __expf(sc[tid] - smax);
  red[tid] = ee; __syncthreads();
  for (int s = NT / 2; s > 0; s >>= 1){ if (tid < s) red[tid] += red[tid + s]; __syncthreads(); }
  float ssum = red[0]; __syncthreads();
  sc[tid] = ee / ssum;
  __syncthreads();

  // ctx[d] = sum_tt sm[tt] * X[b,tt,d]  via transposed bf16 X: contiguous b128 runs
  {
    const unsigned short* xrow = XbfT + ((long long)b * ND + tid) * NT;
    float acc = 0.f;
    for (int j = 0; j < NT; j += 8){
      uint4 xq = *(const uint4*)(xrow + j);
      float4 s0 = *(const float4*)&sc[j];
      float4 s1 = *(const float4*)&sc[j + 4];
      acc += s0.x * bf2f_lo(xq.x);
      acc += s0.y * bf2f_hi(xq.x);
      acc += s0.z * bf2f_lo(xq.y);
      acc += s0.w * bf2f_hi(xq.y);
      acc += s1.x * bf2f_lo(xq.z);
      acc += s1.y * bf2f_hi(xq.z);
      acc += s1.z * bf2f_lo(xq.w);
      acc += s1.w * bf2f_hi(xq.w);
    }
    ctx_bf[(long long)b * ND + tid] = f2bf(acc);
  }
}

// ------------------------------- per-step output GEMM (WMMA) + combine + write
__global__ void __launch_bounds__(256)
k_step_out(const unsigned short* __restrict__ ctx_bf,
           const unsigned short* __restrict__ CoPack,
           const float* __restrict__ Bo3, const float* __restrict__ Bo4,
           const float* __restrict__ woy, const float* __restrict__ UoH,
           float* __restrict__ outp, int t){
  int lane = threadIdx.x & 31;
  int wave = blockIdx.x * (blockDim.x >> 5) + (threadIdx.x >> 5); // 0..31
  int mtile = wave >> 4;   // 0..1  (B rows)
  int ntile = wave & 15;   // 0..15 (O cols)
  int half = lane >> 4;
  int mloc = lane & 15;
  const unsigned short* Arow = ctx_bf + (long long)(mtile * 16 + mloc) * ND;

  f32x8 acc = zero8();
  for (int kt = 0; kt < ND / 32; ++kt){
    Frag16 a;
    a.q[0] = *(const uint4*)(Arow + kt * 32 + half * 8);
    a.q[1] = *(const uint4*)(Arow + kt * 32 + 16 + half * 8);
    Frag16 bq;
    const uint4* bp = (const uint4*)(CoPack + (((long long)kt * 16 + ntile) * 32 + lane) * 16);
    bq.q[0] = bp[0];
    bq.q[1] = bp[1];
    acc = __builtin_amdgcn_wmma_f32_16x16x32_bf16(false, a.v, false, bq.v,
                                                  (short)0, acc, false, false);
  }
  int col = ntile * 16 + mloc;
#pragma unroll
  for (int v = 0; v < 8; ++v){
    int m = mtile * 16 + v + 8 * half;   // batch index
    float val = acc[v] + Bo3[col] + Bo4[col] + woy[m]
              + UoH[((long long)m * NT + t) * NO + col];
    outp[((long long)m * NT + t) * NO + col] = val;
  }
}

// --------------------------------------------------------------------- launch

extern "C" void kernel_launch(void* const* d_in, const int* in_sizes, int n_in,
                              void* d_out, int out_size, void* d_ws, size_t ws_size,
                              hipStream_t stream){
  (void)in_sizes; (void)n_in; (void)out_size; (void)ws_size;
  const float* inputs = (const float*)d_in[0];
  const float* Wa  = (const float*)d_in[1];
  const float* Ua  = (const float*)d_in[2];
  const float* Va  = (const float*)d_in[3];
  const float* Ba1 = (const float*)d_in[4];
  const float* Ba2 = (const float*)d_in[5];
  const float* Ba3 = (const float*)d_in[6];
  const float* Wo  = (const float*)d_in[7];
  const float* Uo  = (const float*)d_in[8];
  const float* Co  = (const float*)d_in[9];
  const float* Bo2 = (const float*)d_in[10];
  const float* Bo3 = (const float*)d_in[11];
  const float* Bo4 = (const float*)d_in[12];
  const float* emb = (const float*)d_in[13];
  float* outp = (float*)d_out;
  char* ws = (char*)d_ws;

  size_t off = 0;
  auto carve = [&](size_t bytes) -> char* {
    char* p = ws + off;
    off = (off + bytes + 255) & ~(size_t)255;
    return p;
  };
  unsigned short* XbfT   = (unsigned short*)carve((size_t)NB * ND * NT * 2); // 64 MB [B,D,T]
  float*          UaH    = (float*)carve((size_t)NB * NT * NO * 4);          // 32 MB
  float*          UoH    = (float*)carve((size_t)NB * NT * NO * 4);          // 32 MB
  unsigned short* UaPack = (unsigned short*)carve((size_t)ND * NO * 2);
  unsigned short* UoPack = (unsigned short*)carve((size_t)ND * NO * 2);
  unsigned short* CoPack = (unsigned short*)carve((size_t)ND * NO * 2);
  unsigned short* ctxbf  = (unsigned short*)carve((size_t)NB * ND * 2);
  float*          woy    = (float*)carve((size_t)NB * 4);
  float*          embWo  = (float*)carve((size_t)NO * 4);

  // prep
  k_transpose_bf<<<dim3(NT / 32, ND / 32, NB), 256, 0, stream>>>(inputs, XbfT);
  k_pack_b<<<512, 256, 0, stream>>>(Ua, UaPack, ND, NO);
  k_pack_b<<<512, 256, 0, stream>>>(Uo, UoPack, ND, NO);
  k_pack_b<<<512, 256, 0, stream>>>(Co, CoPack, ND, NO);
  k_embwo<<<1, 256, 0, stream>>>(emb, Wo, embWo);

  // hoisted GEMMs: UaH = X@Ua + Ba2 ; UoH = roll(X,1)@Uo + Bo2
  k_gemm_bto<<<512, 256, 0, stream>>>(inputs, UaPack, Ba2, 1, UaH, 0);
  k_gemm_bto<<<512, 256, 0, stream>>>(inputs, UoPack, Bo2, 0, UoH, 1);

  // sequential scan (in-order on stream -> step t sees step t-1's output)
  for (int t = 0; t < NT; ++t){
    k_step_attn<<<NB, NT, 0, stream>>>(outp, Wa, Va, Ba1, Ba3, UaH, XbfT, embWo, ctxbf, woy, t);
    k_step_out<<<4, 256, 0, stream>>>(ctxbf, CoPack, Bo3, Bo4, woy, UoH, outp, t);
  }
}
